// G2Conv2d4_19902878449859
// MI455X (gfx1250) — matrix operand order repaired and checked by reading, compiled, not run on hardware
//
#include <hip/hip_runtime.h>

// ---------------------------------------------------------------------------
// Types for CDNA5 WMMA (wave32)
// ---------------------------------------------------------------------------
typedef __attribute__((ext_vector_type(16))) __bf16 v16bf;
typedef __attribute__((ext_vector_type(8)))  float  v8f;

static __device__ __forceinline__ unsigned short f2bf(float f) {
  union { float f; unsigned u; } a; a.f = f;
  unsigned r = a.u + 0x7FFFu + ((a.u >> 16) & 1u);  // round-to-nearest-even
  return (unsigned short)(r >> 16);
}

// ---------------------------------------------------------------------------
// Kernel 1: pack w1,w2 (f32 [128][128]) into the WMMA bf16 A-fragment layout.
// A (16x32, bf16): lanes 0-15 -> M rows, lane/16 selects K-half; per-lane 16
// halves map K = kc*32 + (h>>3)*16 + (lane>>4)*8 + (h&7).
// dst index = (((mt*4 + kc)*32 + lane)*16 + h); 16384 halves per matrix.
// ---------------------------------------------------------------------------
__global__ __launch_bounds__(256) void pack_w_kernel(
    const float* __restrict__ w1, const float* __restrict__ w2,
    unsigned short* __restrict__ dst) {
  int idx = blockIdx.x * 256 + threadIdx.x;        // 0 .. 32767
  const float* w = (idx < 16384) ? w1 : w2;
  int id   = idx & 16383;
  int h    = id & 15;
  int lane = (id >> 4) & 31;
  int kc   = (id >> 9) & 3;
  int mt   = (id >> 11) & 7;
  int m = mt * 16 + (lane & 15);
  int K = kc * 32 + ((h >> 3) << 4) + ((lane >> 4) << 3) + (h & 7);
  dst[idx] = f2bf(w[m * 128 + K]);
}

// ---------------------------------------------------------------------------
// Kernel 2: guided corner pool with orientation folded into addressing.
//   m[i,j] = max(x[i,j], g[i,j]*m[i-1,j], g[i,j]*m[i,j-1]) in oriented coords.
// One wave per (b,c) plane (2048 planes). Lane l owns 4 logical columns;
// 2D wavefront: at step t lane l processes logical row r=t-l; the row carry
// crosses lanes via __shfl_up (computed by lane l-1 one step earlier).
// Prefetch 8 rows ahead to hide HBM latency along the sequential row walk.
// Output written as bf16 (feeds the WMMA GEMM directly).
// ---------------------------------------------------------------------------
__global__ __launch_bounds__(256) void gpool_kernel(
    const float* __restrict__ src, const float* __restrict__ gsrc,
    unsigned short* __restrict__ dst) {
  const int H = 128, W = 128, HW = H * W;
  int wave = threadIdx.x >> 5;
  int lane = threadIdx.x & 31;
  int plane = blockIdx.x * 8 + wave;               // b*C + c, 2048 total
  int c     = plane & 127;
  int chunk = c >> 5;                              // 4 orientation chunks
  bool flipW = (chunk & 1) != 0;
  bool flipH = (chunk & 2) != 0;
  long base = (long)plane * HW;
  int col0 = flipW ? (W - 4 - 4 * lane) : (4 * lane);   // physical float4 base
  int rowstep = flipH ? -W : W;                    // physical stride per logical row

  float mprev0 = 0.f, mprev1 = 0.f, mprev2 = 0.f, mprev3 = 0.f;
  float carry = 0.f;

  for (int t = 0; t < H + 31; ++t) {
    float carry_in = __shfl_up(carry, 1);          // lane l-1's row carry
    int r = t - lane;                              // logical row for this lane
    if (r >= 0 && r < H) {
      int i = flipH ? (H - 1 - r) : r;
      long rowoff = base + (long)i * W + col0;
      // speculative prefetch 8 logical rows ahead (dropped if OOB)
      __builtin_prefetch(src  + rowoff + 8 * rowstep, 0, 0);
      __builtin_prefetch(gsrc + rowoff + 8 * rowstep, 0, 0);
      const float4 xv = *(const float4*)(src  + rowoff);
      const float4 gv = *(const float4*)(gsrc + rowoff);
      float xa[4] = {xv.x, xv.y, xv.z, xv.w};
      float ga[4] = {gv.x, gv.y, gv.z, gv.w};
      float ov[4];
      float mp[4] = {mprev0, mprev1, mprev2, mprev3};
      float cur = carry_in;
      bool haveLeft = (lane != 0);
      bool haveUp = (r > 0);
#pragma unroll
      for (int e = 0; e < 4; ++e) {                // logical column order
        int pe = flipW ? (3 - e) : e;              // physical slot in float4
        float v = xa[pe];
        float g = ga[pe];
        if (haveUp)   v = fmaxf(v, g * mp[e]);
        if (haveLeft) v = fmaxf(v, g * cur);
        haveLeft = true;
        cur = v;
        mp[e] = v;
        ov[pe] = v;
      }
      mprev0 = mp[0]; mprev1 = mp[1]; mprev2 = mp[2]; mprev3 = mp[3];
      carry = cur;
      ushort4 pk;
      pk.x = f2bf(ov[0]); pk.y = f2bf(ov[1]);
      pk.z = f2bf(ov[2]); pk.w = f2bf(ov[3]);
      *(ushort4*)(dst + rowoff) = pk;              // 8-byte store
    }
  }
}

// ---------------------------------------------------------------------------
// Kernel 3: 1x1 conv as bf16 WMMA GEMM: out[o,n] = sum_c w[o,c]*p[c,n] + b[o].
// Block = 8 waves sharing one 16-pixel x 128-channel B tile. The tile is
// staged ONCE per block into LDS with CDNA5 async global->LDS copies
// (ASYNCcnt path), then every wave gathers its B fragments element by element from LDS —
// LDS layout btile[ch*16 + px] is channel-major, so not the B fragment layout
// (there lane%16 = N = pixel, 16 halves = 16 channels in table order).
// Wave mt computes channels 16mt..16mt+15; K=128 in 4 chunks of 32 via
// v_wmma_f32_16x16x32_bf16 with f32 accumulation, then bias add.
// ---------------------------------------------------------------------------
__global__ __launch_bounds__(256) void conv1x1_wmma_kernel(
    const unsigned short* __restrict__ p,      // bf16 activations [B,C,H,W]
    const unsigned short* __restrict__ wpack,  // packed bf16 A fragments
    const float* __restrict__ bias,
    float* __restrict__ out) {
  const int C = 128, HW = 128 * 128;
  __shared__ unsigned short btile[128 * 16];   // 4 KB: [channel][16 pixels]

  int tid  = threadIdx.x;
  int mt   = tid >> 5;                         // wave -> M tile (0..7)
  int lane = tid & 31;
  long n0  = (long)blockIdx.x * 16;            // 16-pixel tile (never crosses b)
  int  b   = (int)(n0 / HW);
  int  pix0 = (int)(n0 % HW);

  // ---- async stage of the shared B tile: 256 threads x 16B = 4KB ----------
  {
    int ch   = tid >> 1;                       // 0..127
    int half = tid & 1;                        // two 16B halves per channel row
    const unsigned short* g = p + ((long)(b * C + ch) * HW + pix0 + half * 8);
    unsigned lds_addr =
        (unsigned)(unsigned long long)(const void*)&btile[ch * 16 + half * 8];
    asm volatile("global_load_async_to_lds_b128 %0, %1, off"
                 :: "v"(lds_addr), "v"((unsigned long long)(const void*)g)
                 : "memory");
  }
  asm volatile("s_wait_asynccnt 0x0" ::: "memory");  // my copies landed in LDS
  __syncthreads();                                   // everyone's copies visible

  v8f acc = {};
#pragma unroll
  for (int kc = 0; kc < 4; ++kc) {
    const v16bf a = *(const v16bf*)(wpack + ((((mt * 4 + kc) * 32) + lane) << 4));
    // B: lane%16 = N (pixel), element s = K (channel) in table order
    v16bf bm;
#pragma unroll
    for (int s = 0; s < 16; ++s) {
      int ch = kc * 32 + ((s >> 3) << 4) + ((lane >> 4) << 3) + (s & 7);
      bm[s] = ((const __bf16*)btile)[ch * 16 + (lane & 15)];
    }
    acc = __builtin_amdgcn_wmma_f32_16x16x32_bf16(
        /*neg_a=*/false, a, /*neg_b=*/false, bm,
        /*c_mod=*/(short)0, acc, /*reuse_a=*/false, /*reuse_b=*/false);
  }

  // D layout: N = lane%16; M = vgpr + 8*(lane/16)
  int n = lane & 15;
  int mhi = (lane >> 4) << 3;
#pragma unroll
  for (int v = 0; v < 8; ++v) {
    int o = mt * 16 + mhi + v;
    out[(long)(b * C + o) * HW + pix0 + n] = acc[v] + bias[o];
  }
}

// ---------------------------------------------------------------------------
// Launch: pack -> pool(x,x) -> gemm(w1) -> pool(c1,x) -> gemm(w2)
// ---------------------------------------------------------------------------
extern "C" void kernel_launch(void* const* d_in, const int* in_sizes, int n_in,
                              void* d_out, int out_size, void* d_ws, size_t ws_size,
                              hipStream_t stream) {
  (void)in_sizes; (void)n_in; (void)out_size; (void)ws_size;
  const float* x  = (const float*)d_in[0];
  const float* w1 = (const float*)d_in[1];
  const float* b1 = (const float*)d_in[2];
  const float* w2 = (const float*)d_in[3];
  const float* b2 = (const float*)d_in[4];
  float* out = (float*)d_out;

  // Workspace layout
  unsigned short* wpack = (unsigned short*)d_ws;                       // 64 KB
  unsigned short* pbuf  = (unsigned short*)((char*)d_ws + 65536);      // 64 MB bf16
  float*          c1    = (float*)((char*)d_ws + 65536 + 67108864);    // 128 MB f32

  pack_w_kernel<<<128, 256, 0, stream>>>(w1, w2, wpack);

  // Layer 1: pool(x, guide=x) -> p ; conv -> c1
  gpool_kernel<<<256, 256, 0, stream>>>(x, x, pbuf);
  conv1x1_wmma_kernel<<<16384, 256, 0, stream>>>(pbuf, wpack, b1, c1);

  // Layer 2: pool(c1, guide=x) -> p ; conv -> out
  gpool_kernel<<<256, 256, 0, stream>>>(c1, x, pbuf);
  conv1x1_wmma_kernel<<<16384, 256, 0, stream>>>(pbuf, wpack + 16384, b2, out);
}
